// MaskedAutoEncoder_89103391523120
// MI455X (gfx1250) — matrix operand order repaired and checked by reading, not compiled, not run on hardware
//
#include <hip/hip_runtime.h>
#include <hip/hip_bf16.h>

#define T_STEPS 20
#define NNODES  50000
#define NEDGES  800000
#define IN_DIM  16
#define HDIM    128
#define NPF     7
#define EPS_LN  1e-5f
#define NTILES  (NNODES / 16)   // 3125, exact

typedef __attribute__((ext_vector_type(16))) __bf16 v16bf;
typedef __attribute__((ext_vector_type(8)))  float  v8f;

// ---------------------------------------------------------------------------
// WMMA helpers (CDNA5 16x16x32 bf16, f32 accumulate)
// ---------------------------------------------------------------------------
__device__ __forceinline__ v8f wmma_bf16(v16bf a, v16bf b, v8f c) {
  return __builtin_amdgcn_wmma_f32_16x16x32_bf16(
      /*neg_a=*/false, a, /*neg_b=*/false, b,
      /*c_mod=*/(short)0, c, /*reuse_a=*/false, /*reuse_b=*/false);
}

// A-matrix 16x32 bf16 tile loader from row-major activations (ISA 7.12.2).
// lane<16:  row=lane,     K in {0..7}  U {16..23}
// lane>=16: row=lane-16,  K in {8..15} U {24..31}
__device__ __forceinline__ v16bf load_a(const __bf16* A, int lda, int lane) {
  int row = lane & 15, kh = (lane >> 4) & 1;
  v16bf a;
#pragma unroll
  for (int v = 0; v < 8; ++v) {
    int k = ((v & 4) << 2) + kh * 8 + ((v & 3) << 1);
    a[2 * v]     = A[(size_t)row * lda + k];
    a[2 * v + 1] = A[(size_t)row * lda + k + 1];
  }
  return a;
}

// Same layout, but source is f32 with a per-row scale (deg_inv fold-in).
__device__ __forceinline__ v16bf load_a_f32s(const float* A, int lda, float scale, int lane) {
  int row = lane & 15, kh = (lane >> 4) & 1;
  v16bf a;
#pragma unroll
  for (int v = 0; v < 8; ++v) {
    int k = ((v & 4) << 2) + kh * 8 + ((v & 3) << 1);
    a[2 * v]     = (__bf16)(A[(size_t)row * lda + k] * scale);
    a[2 * v + 1] = (__bf16)(A[(size_t)row * lda + k + 1] * scale);
  }
  return a;
}

// B-tile loader from swizzled weights: tile = 512 bf16, lane-contiguous 32B.
// Lowers to two global_load_b128 per lane.
__device__ __forceinline__ v16bf load_b_sw(const __bf16* Bsw, int tileIdx, int lane) {
  union { uint4 u[2]; v16bf v; } r;
  const uint4* p = (const uint4*)(Bsw + ((size_t)tileIdx << 9) + (lane << 4));
  r.u[0] = p[0];
  r.u[1] = p[1];
  return r.v;
}

__device__ __forceinline__ float sigmoidf_(float x) { return 1.0f / (1.0f + __expf(-x)); }

__device__ __forceinline__ void store_bf16x2(__bf16* p, float a0, float a1) {
  union { __bf16 h[2]; unsigned int u; } pk;
  pk.h[0] = (__bf16)a0; pk.h[1] = (__bf16)a1;
  *(unsigned int*)p = pk.u;
}

// ---------------------------------------------------------------------------
// Utility kernels
// ---------------------------------------------------------------------------
__global__ void k_zero_f32(float* p, long long n) {
  long long i = (long long)blockIdx.x * blockDim.x + threadIdx.x;
  if (i < n) p[i] = 0.0f;
}
__global__ void k_zero_u16(unsigned short* p, long long n) {
  long long i = (long long)blockIdx.x * blockDim.x + threadIdx.x;
  if (i < n) p[i] = 0;
}
__global__ void k_degree(const int* dst, float* deg, int E) {
  int e = blockIdx.x * blockDim.x + threadIdx.x;
  if (e < E) atomicAdd(&deg[dst[e]], 1.0f);
}
__global__ void k_deg_inv(float* deg, int n) {
  int i = blockIdx.x * blockDim.x + threadIdx.x;
  if (i < n) deg[i] = 1.0f / fmaxf(deg[i], 1.0f);
}

// Swizzle (and bf16-convert) a [K x C] f32 weight into WMMA-B tile layout:
// tile (nt*Kt + kt) holds 32 lanes x 16 bf16 contiguous.  Source may be two
// matrices stacked along K (rows < Ksplit from srcA, rest from srcB) so the
// SAGE [W_l ; W_r] concat is produced directly.
__global__ void k_swizzle_b(const float* srcA, const float* srcB, int Ksplit,
                            int K, int C, __bf16* outSw) {
  long long idx = (long long)blockIdx.x * blockDim.x + threadIdx.x;
  if (idx >= (long long)K * C) return;
  int Kt = K >> 5;
  int tile = (int)(idx >> 9);
  int rem  = (int)(idx & 511);
  int lane = rem >> 4, j = rem & 15;
  int nt = tile / Kt, kt = tile - nt * Kt;
  int k = kt * 32 + ((j >> 3) << 4) + ((lane >> 4) << 3) + (j & 7);
  int c = (lane & 15) + nt * 16;
  float v = (k < Ksplit) ? srcA[(size_t)k * C + c]
                         : srcB[(size_t)(k - Ksplit) * C + c];
  outSw[idx] = (__bf16)v;
}

// Same, but source is [C x K] row-major and we swizzle B = src^T ([K x C]).
__global__ void k_swizzle_bt(const float* src, int K, int C, __bf16* outSw) {
  long long idx = (long long)blockIdx.x * blockDim.x + threadIdx.x;
  if (idx >= (long long)K * C) return;
  int Kt = K >> 5;
  int tile = (int)(idx >> 9);
  int rem  = (int)(idx & 511);
  int lane = rem >> 4, j = rem & 15;
  int nt = tile / Kt, kt = tile - nt * Kt;
  int k = kt * 32 + ((j >> 3) << 4) + ((lane >> 4) << 3) + (j & 7);
  int c = (lane & 15) + nt * 16;
  outSw[idx] = (__bf16)src[(size_t)c * K + k];
}

// Scatter-add of raw x_t (f32): 4 features per thread -> index loads amortized,
// feature read is one float4.  Accumulator is L2-resident (3.2 MB).
__global__ void k_scatter16(const int* __restrict__ src, const int* __restrict__ dst,
                            const float* __restrict__ xt, float* agg, int E) {
  long long idx = (long long)blockIdx.x * blockDim.x + threadIdx.x;
  if (idx >= (long long)E * 4) return;
  int e = (int)(idx >> 2), q = (int)(idx & 3);
  int s = src[e], d = dst[e];
  float4 v = *(const float4*)(xt + (size_t)s * 16 + q * 4);
  float* ap = agg + (size_t)d * 16 + q * 4;
  atomicAdd(ap + 0, v.x);
  atomicAdd(ap + 1, v.y);
  atomicAdd(ap + 2, v.z);
  atomicAdd(ap + 3, v.w);
}

// Scatter-add of h1 (bf16): 4 features per thread, one 8B load of 4 bf16,
// f32 atomics into the L2-resident 25.6 MB accumulator.
__global__ void k_scatter128(const int* __restrict__ src, const int* __restrict__ dst,
                             const __bf16* __restrict__ h, float* agg, int E) {
  long long idx = (long long)blockIdx.x * blockDim.x + threadIdx.x;
  if (idx >= (long long)E * 32) return;
  int e = (int)(idx >> 5), q = (int)(idx & 31);
  int s = src[e], d = dst[e];
  union { uint2 u; __bf16 h[4]; } v;
  v.u = *(const uint2*)(h + (size_t)s * HDIM + q * 4);
  float* ap = agg + (size_t)d * HDIM + q * 4;
  atomicAdd(ap + 0, (float)v.h[0]);
  atomicAdd(ap + 1, (float)v.h[1]);
  atomicAdd(ap + 2, (float)v.h[2]);
  atomicAdd(ap + 3, (float)v.h[3]);
}

// ---------------------------------------------------------------------------
// SAGE1: h1 = relu([agg*dinv ; x_t](16x32) @ [wl;wr](32x128) + bl), bf16 out
// one wave per 16-row M-tile; 256-thread blocks = 8 waves
// ---------------------------------------------------------------------------
__global__ void k_sage1(const float* agg1, const float* deg_inv, const float* xt,
                        const __bf16* w1sw, const float* bl, __bf16* h1) {
  int lane = threadIdx.x & 31;
  int tile = blockIdx.x * (blockDim.x >> 5) + (threadIdx.x >> 5);
  if (tile >= NTILES) return;
  int kh = (lane >> 4) & 1, colL = lane & 15;
  int rowA = tile * 16 + colL;
  float dinv = deg_inv[rowA];

  // prefetch next tile's A-operand streams (global_prefetch_b8)
  if (tile + 1 < NTILES) {
    __builtin_prefetch((const char*)(agg1 + (size_t)(tile + 1) * 16 * IN_DIM) + lane * 32, 0, 1);
    __builtin_prefetch((const char*)(xt + (size_t)(tile + 1) * 16 * IN_DIM) + lane * 32, 0, 1);
  }

  v16bf a;
#pragma unroll
  for (int v = 0; v < 4; ++v) {  // K=0..15: mean-aggregated neighbors
    int k = kh * 8 + 2 * v;
    a[2 * v]     = (__bf16)(agg1[(size_t)rowA * 16 + k] * dinv);
    a[2 * v + 1] = (__bf16)(agg1[(size_t)rowA * 16 + k + 1] * dinv);
  }
#pragma unroll
  for (int v = 4; v < 8; ++v) {  // K=16..31: root features x_t
    int k = kh * 8 + 2 * (v - 4);
    a[2 * v]     = (__bf16)xt[(size_t)rowA * 16 + k];
    a[2 * v + 1] = (__bf16)xt[(size_t)rowA * 16 + k + 1];
  }

#pragma unroll
  for (int nt = 0; nt < 8; ++nt) {
    v8f c = {};
    c = wmma_bf16(a, load_b_sw(w1sw, nt, lane), c);  // Kt = 1
    int colg = nt * 16 + colL;
    float bias = bl[colg];
#pragma unroll
    for (int v = 0; v < 8; ++v) {
      int rg = tile * 16 + kh * 8 + v;
      h1[(size_t)rg * HDIM + colg] = (__bf16)fmaxf(c[v] + bias, 0.0f);
    }
  }
}

// ---------------------------------------------------------------------------
// SAGE2 + fused LayerNorm: enc = LN(relu([agg2*dinv ; h1](16x256) @ W2cat))
// one wave per M-tile; 128-thread blocks = 4 waves; wave-private 16x128 LDS
// ---------------------------------------------------------------------------
__global__ void k_sage2_ln(const float* agg2, const float* deg_inv, const __bf16* h1,
                           const __bf16* w2sw, const float* bl,
                           const float* lng, const float* lnb, __bf16* enc) {
  __shared__ float lds[4][16][HDIM];  // 32 KB
  int lane = threadIdx.x & 31, wave = threadIdx.x >> 5;
  int tile = blockIdx.x * 4 + wave;
  if (tile >= NTILES) return;
  int kh = (lane >> 4) & 1, colL = lane & 15;
  int rowA = tile * 16 + colL;
  float dinv = deg_inv[rowA];

  // prefetch next tile's activation streams while this tile computes
  if (tile + 4 < NTILES) {  // next tile handled by this wave-slot's neighbor grid step
    __builtin_prefetch((const char*)(agg2 + (size_t)(tile + 1) * 16 * HDIM) + lane * 256, 0, 1);
    __builtin_prefetch((const char*)(h1 + (size_t)(tile + 1) * 16 * HDIM) + lane * 128, 0, 1);
  }

  v16bf at[8];
#pragma unroll
  for (int kt = 0; kt < 4; ++kt) {  // K 0..127: mean-aggregated neighbors
    at[kt] = load_a_f32s(agg2 + (size_t)tile * 16 * HDIM + kt * 32, HDIM, dinv, lane);
  }
#pragma unroll
  for (int kt = 0; kt < 4; ++kt) {  // K 128..255: root h1
    at[kt + 4] = load_a(h1 + (size_t)tile * 16 * HDIM + kt * 32, HDIM, lane);
  }

#pragma unroll
  for (int nt = 0; nt < 8; ++nt) {
    v8f c = {};
#pragma unroll
    for (int kt = 0; kt < 8; ++kt) {
      c = wmma_bf16(at[kt], load_b_sw(w2sw, nt * 8 + kt, lane), c);  // Kt = 8
    }
    int colg = nt * 16 + colL;
    float bias = bl[colg];
#pragma unroll
    for (int v = 0; v < 8; ++v) {
      lds[wave][kh * 8 + v][colg] = fmaxf(c[v] + bias, 0.0f);
    }
  }
  asm volatile("s_wait_dscnt 0" ::: "memory");  // wave-private slab, DS in-order

  // 32-lane LayerNorm: lane owns row=lane&15, half kh covers 64 columns;
  // cross-half combine via shfl_xor(16).
  {
    int row = lane & 15, rg = tile * 16 + row;
    const float* rp = lds[wave][row];
    int c0 = kh * 64;
    float s = 0.0f;
#pragma unroll 8
    for (int k = 0; k < 64; ++k) s += rp[c0 + k];
    s += __shfl_xor(s, 16, 32);
    float mu = s * (1.0f / HDIM);
    float vv = 0.0f;
#pragma unroll 8
    for (int k = 0; k < 64; ++k) { float d = rp[c0 + k] - mu; vv += d * d; }
    vv += __shfl_xor(vv, 16, 32);
    float rs = rsqrtf(vv * (1.0f / HDIM) + EPS_LN);
#pragma unroll 8
    for (int k = 0; k < 64; k += 2) {
      float a0 = (rp[c0 + k]     - mu) * rs * lng[c0 + k]     + lnb[c0 + k];
      float a1 = (rp[c0 + k + 1] - mu) * rs * lng[c0 + k + 1] + lnb[c0 + k + 1];
      store_bf16x2(enc + (size_t)rg * HDIM + c0 + k, a0, a1);
    }
  }
}

// ---------------------------------------------------------------------------
// GRU step, fully fused in registers: per 16-col tile accumulate r/z/n gates
// for both GEMMs (enc@Wih^T, h@Whh^T), then gate math -> h_new (bf16).
// 128-thread blocks = 4 waves, one M-tile per wave.
// ---------------------------------------------------------------------------
__global__ void k_gru(const __bf16* enc, const __bf16* hprev,
                      const __bf16* wih_sw, const __bf16* whh_sw,  // swizzled 128x384
                      const float* bih, const float* bhh, __bf16* hnew) {
  int lane = threadIdx.x & 31;
  int tile = blockIdx.x * (blockDim.x >> 5) + (threadIdx.x >> 5);
  if (tile >= NTILES) return;
  int kh = (lane >> 4) & 1, colL = lane & 15;
  size_t base = (size_t)tile * 16 * HDIM;

  if (tile + 1 < NTILES) {
    __builtin_prefetch((const char*)(enc + base + 16 * HDIM) + lane * 128, 0, 1);
    __builtin_prefetch((const char*)(hprev + base + 16 * HDIM) + lane * 128, 0, 1);
  }

  v16bf ae[4], ah[4];
#pragma unroll
  for (int kt = 0; kt < 4; ++kt) {
    ae[kt] = load_a(enc + base + kt * 32, HDIM, lane);
    ah[kt] = load_a(hprev + base + kt * 32, HDIM, lane);
  }

#pragma unroll
  for (int nt = 0; nt < 8; ++nt) {
    v8f ar = {}, az = {}, ain = {}, ahn = {};
#pragma unroll
    for (int kt = 0; kt < 4; ++kt) {
      // Kt=4, Ct=24; column-tile index = (section*8 + nt)
      ar  = wmma_bf16(ae[kt], load_b_sw(wih_sw, (nt)      * 4 + kt, lane), ar);
      ar  = wmma_bf16(ah[kt], load_b_sw(whh_sw, (nt)      * 4 + kt, lane), ar);
      az  = wmma_bf16(ae[kt], load_b_sw(wih_sw, (8 + nt)  * 4 + kt, lane), az);
      az  = wmma_bf16(ah[kt], load_b_sw(whh_sw, (8 + nt)  * 4 + kt, lane), az);
      ain = wmma_bf16(ae[kt], load_b_sw(wih_sw, (16 + nt) * 4 + kt, lane), ain);
      ahn = wmma_bf16(ah[kt], load_b_sw(whh_sw, (16 + nt) * 4 + kt, lane), ahn);
    }
    int colg = nt * 16 + colL;
    float br  = bih[colg]       + bhh[colg];
    float bz  = bih[128 + colg] + bhh[128 + colg];
    float bin = bih[256 + colg];
    float bhn = bhh[256 + colg];
#pragma unroll
    for (int v = 0; v < 8; ++v) {
      int rg = tile * 16 + kh * 8 + v;
      float r = sigmoidf_(ar[v] + br);
      float z = sigmoidf_(az[v] + bz);
      float n = tanhf((ain[v] + bin) + r * (ahn[v] + bhn));
      float hp = (float)hprev[(size_t)rg * HDIM + colg];
      hnew[(size_t)rg * HDIM + colg] = (__bf16)((1.0f - z) * n + z * hp);
    }
  }
}

// ---------------------------------------------------------------------------
// Decoder: out = relu(h @ W1 + b1) @ W2 + b2   (W2: 128 x OD, OD = 1 or 7)
// hidden tile via WMMA -> wave-private LDS -> 32-lane dot for OD outputs.
// ---------------------------------------------------------------------------
__global__ void k_decoder(const __bf16* h, const __bf16* w1sw, const float* b1,
                          const float* w2, const float* b2, float* out, int OD) {
  __shared__ float lds[4][16][HDIM];  // 32 KB
  int lane = threadIdx.x & 31, wave = threadIdx.x >> 5;
  int tile = blockIdx.x * 4 + wave;
  if (tile >= NTILES) return;
  int kh = (lane >> 4) & 1, colL = lane & 15;
  size_t base = (size_t)tile * 16 * HDIM;

  if (tile + 1 < NTILES) {
    __builtin_prefetch((const char*)(h + base + 16 * HDIM) + lane * 128, 0, 1);
  }

  v16bf at[4];
#pragma unroll
  for (int kt = 0; kt < 4; ++kt) at[kt] = load_a(h + base + kt * 32, HDIM, lane);

#pragma unroll
  for (int nt = 0; nt < 8; ++nt) {
    v8f c = {};
#pragma unroll
    for (int kt = 0; kt < 4; ++kt) {
      c = wmma_bf16(at[kt], load_b_sw(w1sw, nt * 4 + kt, lane), c);  // Kt = 4
    }
    int colg = nt * 16 + colL;
    float bias = b1[colg];
#pragma unroll
    for (int v = 0; v < 8; ++v) {
      lds[wave][kh * 8 + v][colg] = fmaxf(c[v] + bias, 0.0f);
    }
  }
  asm volatile("s_wait_dscnt 0" ::: "memory");

  // 32-lane second layer: lane owns row=lane&15 over 64 cols; shfl combine.
  {
    int row = lane & 15, rg = tile * 16 + row;
    const float* rp = lds[wave][row];
    int c0 = kh * 64;
    for (int j = 0; j < OD; ++j) {
      float s = 0.0f;
#pragma unroll 8
      for (int k = 0; k < 64; ++k) s += rp[c0 + k] * w2[(size_t)(c0 + k) * OD + j];
      s += __shfl_xor(s, 16, 32);
      if (kh == 0) out[(size_t)rg * OD + j] = s + b2[j];
    }
  }
}

// ---------------------------------------------------------------------------
// Host orchestration
// ---------------------------------------------------------------------------
static inline int ceil_div(long long a, int b) { return (int)((a + b - 1) / b); }

extern "C" void kernel_launch(void* const* d_in, const int* in_sizes, int n_in,
                              void* d_out, int out_size, void* d_ws, size_t ws_size,
                              hipStream_t stream) {
  const float* x      = (const float*)d_in[0];
  const int*   ei     = (const int*)d_in[1];      // (2, E): row0=src, row1=dst
  const float* s1wl   = (const float*)d_in[2];
  const float* s1bl   = (const float*)d_in[3];
  const float* s1wr   = (const float*)d_in[4];
  const float* s2wl   = (const float*)d_in[5];
  const float* s2bl   = (const float*)d_in[6];
  const float* s2wr   = (const float*)d_in[7];
  const float* lng    = (const float*)d_in[8];
  const float* lnb    = (const float*)d_in[9];
  const float* g_wih  = (const float*)d_in[10];   // [384,128]
  const float* g_whh  = (const float*)d_in[11];
  const float* g_bih  = (const float*)d_in[12];
  const float* g_bhh  = (const float*)d_in[13];
  const float* th_w1  = (const float*)d_in[14];
  const float* th_b1  = (const float*)d_in[15];
  const float* th_w2  = (const float*)d_in[16];
  const float* th_b2  = (const float*)d_in[17];
  const float* ph_w1  = (const float*)d_in[18];
  const float* ph_b1  = (const float*)d_in[19];
  const float* ph_w2  = (const float*)d_in[20];
  const float* ph_b2  = (const float*)d_in[21];
  float* out = (float*)d_out;

  const int* src = ei;
  const int* dst = ei + NEDGES;

  // ---- workspace carve ----
  char* ws = (char*)d_ws;
  auto carve = [&](size_t bytes) { char* p = ws; ws += (bytes + 255) & ~(size_t)255; return p; };
  float*  deg    = (float*)carve((size_t)NNODES * 4);            // -> deg_inv in place
  float*  agg1   = (float*)carve((size_t)NNODES * IN_DIM * 4);
  float*  agg2   = (float*)carve((size_t)NNODES * HDIM * 4);
  __bf16* h1     = (__bf16*)carve((size_t)NNODES * HDIM * 2);
  __bf16* enc    = (__bf16*)carve((size_t)NNODES * HDIM * 2);
  __bf16* hA     = (__bf16*)carve((size_t)NNODES * HDIM * 2);
  __bf16* hB     = (__bf16*)carve((size_t)NNODES * HDIM * 2);
  __bf16* w1sw   = (__bf16*)carve((size_t)32  * HDIM * 2);       // [wl;wr] swizzled
  __bf16* w2sw   = (__bf16*)carve((size_t)256 * HDIM * 2);       // [wl;wr] swizzled
  __bf16* wih_sw = (__bf16*)carve((size_t)HDIM * 384 * 2);       // Wih^T swizzled
  __bf16* whh_sw = (__bf16*)carve((size_t)HDIM * 384 * 2);       // Whh^T swizzled
  __bf16* thw1sw = (__bf16*)carve((size_t)HDIM * HDIM * 2);
  __bf16* phw1sw = (__bf16*)carve((size_t)HDIM * HDIM * 2);

  // ---- setup: degree, weight prepack (swizzle), h0 = 0 ----
  k_zero_f32<<<ceil_div(NNODES, 256), 256, 0, stream>>>(deg, NNODES);
  k_degree<<<ceil_div(NEDGES, 256), 256, 0, stream>>>(dst, deg, NEDGES);
  k_deg_inv<<<ceil_div(NNODES, 256), 256, 0, stream>>>(deg, NNODES);

  k_swizzle_b<<<ceil_div(32 * HDIM, 256), 256, 0, stream>>>(s1wl, s1wr, 16, 32, HDIM, w1sw);
  k_swizzle_b<<<ceil_div(256 * HDIM, 256), 256, 0, stream>>>(s2wl, s2wr, 128, 256, HDIM, w2sw);
  k_swizzle_b<<<ceil_div(HDIM * HDIM, 256), 256, 0, stream>>>(th_w1, th_w1, HDIM, HDIM, HDIM, thw1sw);
  k_swizzle_b<<<ceil_div(HDIM * HDIM, 256), 256, 0, stream>>>(ph_w1, ph_w1, HDIM, HDIM, HDIM, phw1sw);
  k_swizzle_bt<<<ceil_div(HDIM * 384, 256), 256, 0, stream>>>(g_wih, HDIM, 384, wih_sw);
  k_swizzle_bt<<<ceil_div(HDIM * 384, 256), 256, 0, stream>>>(g_whh, HDIM, 384, whh_sw);
  k_zero_u16<<<ceil_div((long long)NNODES * HDIM, 256), 256, 0, stream>>>(
      (unsigned short*)hA, (long long)NNODES * HDIM);

  __bf16* hprev = hA;
  __bf16* hcur  = hB;

  const int g_sage1 = ceil_div(NTILES, 8);   // 256-thread blocks, 8 waves
  const int g_w4    = ceil_div(NTILES, 4);   // 128-thread blocks, 4 waves

  for (int t = 0; t < T_STEPS; ++t) {
    const float* xt = x + (size_t)t * NNODES * IN_DIM;

    // SAGE layer 1
    k_zero_f32<<<ceil_div((long long)NNODES * IN_DIM, 256), 256, 0, stream>>>(
        agg1, (long long)NNODES * IN_DIM);
    k_scatter16<<<ceil_div((long long)NEDGES * 4, 256), 256, 0, stream>>>(
        src, dst, xt, agg1, NEDGES);
    k_sage1<<<g_sage1, 256, 0, stream>>>(agg1, deg, xt, w1sw, s1bl, h1);

    // SAGE layer 2 + LayerNorm
    k_zero_f32<<<ceil_div((long long)NNODES * HDIM, 256), 256, 0, stream>>>(
        agg2, (long long)NNODES * HDIM);
    k_scatter128<<<ceil_div((long long)NEDGES * 32, 256), 256, 0, stream>>>(
        src, dst, h1, agg2, NEDGES);
    k_sage2_ln<<<g_w4, 128, 0, stream>>>(agg2, deg, h1, w2sw, s2bl, lng, lnb, enc);

    // GRU step (register-fused gates)
    k_gru<<<g_w4, 128, 0, stream>>>(enc, hprev, wih_sw, whh_sw, g_bih, g_bhh, hcur);

    // Decoders on h_t: th -> out[0 .. T*N), ph -> out[T*N .. T*N + T*N*7)
    k_decoder<<<g_w4, 128, 0, stream>>>(hcur, thw1sw, th_b1, th_w2, th_b2,
                                        out + (size_t)t * NNODES, 1);
    k_decoder<<<g_w4, 128, 0, stream>>>(hcur, phw1sw, ph_b1, ph_w2, ph_b2,
                                        out + (size_t)T_STEPS * NNODES + (size_t)t * NNODES * NPF, NPF);

    // ping-pong hidden state
    __bf16* tmp = hprev; hprev = hcur; hcur = tmp;
  }
}
